// TopKMoE_3908420239761
// MI455X (gfx1250) — compile-verified, matrix operand order checked by reading, not run on hardware
//
#include <hip/hip_runtime.h>

// ---------------- problem constants ----------------
#define DM    1024          // d_model
#define DF    4096          // d_ff
#define NE    8             // experts
#define TOPK  2
#define T_TOK 8192          // 4*2048 tokens
#define NTOT  (T_TOK*TOPK)  // 16384 dispatch entries
#define CAP   2048          // ceil(1.0 * NTOT / NE)
#define ECAP  (NE*CAP)      // 16384 expert-buffer rows

typedef __attribute__((ext_vector_type(16))) __bf16 v16bf;
typedef __attribute__((ext_vector_type(8)))  float  v8f;

// WMMA 16x16x32 bf16 fragment K-swizzle (A: lane=row, B: lane=col)
__device__ __forceinline__ int frag_k(int j, int half) {
    int v = j >> 1, p = j & 1;
    return (((v & 3) << 1) | p) + (half << 3) + ((v >> 2) << 4);
}

// CDNA5 async memory->LDS DMA (ASYNCcnt-tracked), ISA §15.18.3 opcode 98.
__device__ __forceinline__ void async_load_b128(unsigned lds_addr, const void* gptr) {
    asm volatile("global_load_async_to_lds_b128 %0, %1, off"
                 :: "v"(lds_addr), "v"((unsigned long long)(uintptr_t)gptr)
                 : "memory");
}
__device__ __forceinline__ void wait_async0() {
    asm volatile("s_wait_asynccnt 0x0" ::: "memory");
}

// pack two floats into packed bf16 pair (lowers to v_cvt_pk_bf16_f32)
__device__ __forceinline__ unsigned pack_bf16(float lo, float hi) {
    __bf16 l = (__bf16)lo, h = (__bf16)hi;
    unsigned short ul = __builtin_bit_cast(unsigned short, l);
    unsigned short uh = __builtin_bit_cast(unsigned short, h);
    return (unsigned)ul | ((unsigned)uh << 16);
}

// ---------------- 0: init scatter targets ----------------
__global__ __launch_bounds__(256)
void init_kernel(int* __restrict__ inv_slot) {
    int i = blockIdx.x * 256 + threadIdx.x;
    if (i < ECAP) inv_slot[i] = -1;
}

// ---------------- 1: router (logits -> softmax -> top2 -> gates) ----------------
__global__ __launch_bounds__(256)
void router_kernel(const float* __restrict__ x, const float* __restrict__ rw,
                   const float* __restrict__ rb,
                   float* __restrict__ gate, int* __restrict__ eidx,
                   float* __restrict__ blkImp, int* __restrict__ blkLoad) {
    __shared__ float sW[DM * NE];       // 32 KB router weights
    __shared__ float sP[256][NE];       //  8 KB per-thread probs (for det. reduce)
    __shared__ int   sLoad[NE];

    const int tid = threadIdx.x;
    for (int i = tid; i < DM * NE; i += 256) sW[i] = rw[i];
    if (tid < NE) sLoad[tid] = 0;
    __syncthreads();

    const int t = blockIdx.x * 256 + tid;      // one token per thread (32 blocks)
    float logit[NE];
    #pragma unroll
    for (int e = 0; e < NE; ++e) logit[e] = rb[e];
    const float* xr = x + (size_t)t * DM;
    for (int d = 0; d < DM; ++d) {
        float xv = xr[d];
        #pragma unroll
        for (int e = 0; e < NE; ++e) logit[e] += xv * sW[d * NE + e];
    }
    // softmax (f32)
    float mx = logit[0];
    #pragma unroll
    for (int e = 1; e < NE; ++e) mx = fmaxf(mx, logit[e]);
    float p[NE], sum = 0.f;
    #pragma unroll
    for (int e = 0; e < NE; ++e) { p[e] = __expf(logit[e] - mx); sum += p[e]; }
    float inv = 1.f / sum;
    #pragma unroll
    for (int e = 0; e < NE; ++e) p[e] *= inv;
    // top-2 (lowest index wins ties, like lax.top_k)
    int i1 = 0;
    #pragma unroll
    for (int e = 1; e < NE; ++e) if (p[e] > p[i1]) i1 = e;
    int i2 = (i1 == 0) ? 1 : 0;
    #pragma unroll
    for (int e = 0; e < NE; ++e) if (e != i1 && p[e] > p[i2]) i2 = e;
    float s2 = fmaxf(p[i1] + p[i2], 1e-9f);
    gate[2 * t]     = p[i1] / s2;
    gate[2 * t + 1] = p[i2] / s2;
    eidx[2 * t]     = i1;
    eidx[2 * t + 1] = i2;

    // deterministic per-block reductions for aux loss
    #pragma unroll
    for (int e = 0; e < NE; ++e) sP[tid][e] = p[e];
    atomicAdd(&sLoad[i1], 1);                   // int add: exact/commutative
    __syncthreads();
    if (tid < NE) {
        float s = 0.f;
        for (int i = 0; i < 256; ++i) s += sP[i][tid];   // fixed order
        blkImp[blockIdx.x * NE + tid]  = s;
        blkLoad[blockIdx.x * NE + tid] = sLoad[tid];
    }
}

// ---------------- 2: stable rank-within-expert + capacity drop ----------------
__global__ __launch_bounds__(256)
void scan_kernel(const int* __restrict__ eidx, int* __restrict__ slot,
                 int* __restrict__ inv_slot) {
    __shared__ int sF[256];
    const int e = blockIdx.x;                   // one block per expert
    const int tid = threadIdx.x;
    int base = 0;
    for (int c = 0; c < NTOT; c += 256) {
        const int n = c + tid;
        const int f = (eidx[n] == e) ? 1 : 0;
        sF[tid] = f;
        __syncthreads();
        for (int off = 1; off < 256; off <<= 1) {   // Hillis-Steele inclusive scan
            int v = (tid >= off) ? sF[tid - off] : 0;
            __syncthreads();
            sF[tid] += v;
            __syncthreads();
        }
        if (f) {
            int pos = base + sF[tid] - 1;           // stable rank within expert
            int s = (pos < CAP) ? (e * CAP + pos) : ECAP;   // ECAP == dump row
            slot[n] = s;
            if (pos < CAP) inv_slot[s] = n;
        }
        base += sF[255];
        __syncthreads();
    }
}

// ---------------- 3: gather tokens into bf16 expert buffers ----------------
__global__ __launch_bounds__(256)
void gather_kernel(const float* __restrict__ x, const int* __restrict__ inv_slot,
                   __bf16* __restrict__ buf) {
    const int r = blockIdx.x;                   // ECAP rows
    const int n = inv_slot[r];
    unsigned* dst = (unsigned*)(buf + (size_t)r * DM);   // row is 4B-aligned
    const int tid = threadIdx.x;
    if (n >= 0) {
        const float2* src = (const float2*)(x + (size_t)(n >> 1) * DM);
        #pragma unroll
        for (int it = 0; it < 2; ++it) {            // 512 pairs / 256 threads
            float2 v = src[tid + it * 256];
            dst[tid + it * 256] = pack_bf16(v.x, v.y);
        }
    } else {
        #pragma unroll
        for (int it = 0; it < 2; ++it) dst[tid + it * 256] = 0u;
    }
}

// ---------------- 4: GEMM1  h = silu(X@W1) * (X@W3)  (bf16 WMMA, f32 acc) ----------------
__global__ __launch_bounds__(256)
void gemm1_kernel(const __bf16* __restrict__ buf,   // [E,CAP,DM] bf16
                  const float* __restrict__ w1,     // [E,DM,DF] f32
                  const float* __restrict__ w3,     // [E,DM,DF] f32
                  __bf16* __restrict__ h) {         // [E,CAP,DF] bf16
    __shared__ __bf16 sA[64][32];                   // X tile   (row-major)
    __shared__ __bf16 sB1[128][32];                 // W1 tile  ([f][k])
    __shared__ __bf16 sB3[128][32];                 // W3 tile  ([f][k])

    const int e     = blockIdx.z;
    const int m_blk = blockIdx.y * 64;              // over CAP
    const int n_blk = blockIdx.x * 128;             // over DF

    const __bf16* A  = buf + (size_t)e * CAP * DM;
    const float*  B1 = w1  + (size_t)e * DM * DF;
    const float*  B3 = w3  + (size_t)e * DM * DF;

    const int tid  = threadIdx.x;
    const int lane = tid & 31;
    const int half = lane >> 4;
    const int l16  = lane & 15;
    const int wm   = (tid >> 5) >> 2;               // 0..1  (M)
    const int wf   = (tid >> 5) & 3;                // 0..3  (F)

    // per-thread chunk for the async A-tile DMA: 64x32 bf16 = 256 x 16B
    const int a_row = tid >> 2;
    const int a_col = (tid & 3) * 8;                // 8 bf16 = 16 B
    const unsigned a_lds = (unsigned)(uintptr_t)&sA[a_row][a_col];

    v8f acc1[2][2] = {};
    v8f acc3[2][2] = {};

    for (int k0 = 0; k0 < DM; k0 += 32) {
        __syncthreads();                            // previous tile fully consumed
        // X tile: async DMA memory->LDS (ASYNCcnt); overlaps the weight staging
        async_load_b128(a_lds, &A[(size_t)(m_blk + a_row) * DM + (k0 + a_col)]);
        // W tiles: fp32 -> bf16, 2(k)x2(f) micro-tile per thread:
        // 2x global b64 loads, packed cvt, 1x ds_store_b32 per column
        #pragma unroll
        for (int it = 0; it < 4; ++it) {            // 32*128/4 = 1024 quads
            int i  = tid + it * 256;
            int kk = (i >> 6) * 2;                  // 0,2,..,30
            int ff = (i & 63) * 2;                  // 0,2,..,126
            const float2* p1a = (const float2*)&B1[(size_t)(k0 + kk)     * DF + (n_blk + ff)];
            const float2* p1b = (const float2*)&B1[(size_t)(k0 + kk + 1) * DF + (n_blk + ff)];
            const float2* p3a = (const float2*)&B3[(size_t)(k0 + kk)     * DF + (n_blk + ff)];
            const float2* p3b = (const float2*)&B3[(size_t)(k0 + kk + 1) * DF + (n_blk + ff)];
            float2 a1 = *p1a, b1v = *p1b, a3 = *p3a, b3v = *p3b;
            *(unsigned*)&sB1[ff][kk]     = pack_bf16(a1.x, b1v.x);
            *(unsigned*)&sB1[ff + 1][kk] = pack_bf16(a1.y, b1v.y);
            *(unsigned*)&sB3[ff][kk]     = pack_bf16(a3.x, b3v.x);
            *(unsigned*)&sB3[ff + 1][kk] = pack_bf16(a3.y, b3v.y);
        }
        // pull next k-chunk weight rows toward L2 (global_prefetch_b8)
        if (k0 + 32 < DM) {
            int p  = tid & 127;
            int kk = p & 31, ff = (p >> 5) * 32;    // 128B cacheline stride
            const float* base = (tid < 128) ? B1 : B3;
            __builtin_prefetch(&base[(size_t)(k0 + 32 + kk) * DF + (n_blk + ff)], 0, 1);
        }
        wait_async0();                              // this wave's DMA landed in LDS
        __syncthreads();                            // all waves' tiles visible

        #pragma unroll
        for (int i = 0; i < 2; ++i) {
            v16bf af;
            const int ar = wm * 32 + i * 16 + l16;
            #pragma unroll
            for (int j = 0; j < 16; ++j) af[j] = sA[ar][frag_k(j, half)];
            #pragma unroll
            for (int jj = 0; jj < 2; ++jj) {
                const int bc = wf * 32 + jj * 16 + l16;
                v16bf b1, b3;
                #pragma unroll
                for (int j = 0; j < 16; ++j) {
                    int kk = frag_k(j, half);
                    b1[j] = sB1[bc][kk];
                    b3[j] = sB3[bc][kk];
                }
                acc1[i][jj] = __builtin_amdgcn_wmma_f32_16x16x32_bf16(
                    false, af, false, b1, (short)0, acc1[i][jj], false, false);
                acc3[i][jj] = __builtin_amdgcn_wmma_f32_16x16x32_bf16(
                    false, af, false, b3, (short)0, acc3[i][jj], false, false);
            }
        }
    }

    // fused SwiGLU epilogue: h = silu(a1) * a3
    __bf16* H = h + (size_t)e * CAP * DF;
    #pragma unroll
    for (int i = 0; i < 2; ++i)
        #pragma unroll
        for (int jj = 0; jj < 2; ++jj)
            #pragma unroll
            for (int r = 0; r < 8; ++r) {
                int row = m_blk + wm * 32 + i * 16 + r + (half << 3);
                int col = n_blk + wf * 32 + jj * 16 + l16;
                float a = acc1[i][jj][r];
                float s = a / (1.0f + __expf(-a));
                H[(size_t)row * DF + col] = (__bf16)(s * acc3[i][jj][r]);
            }
}

// ---------------- 5: GEMM2  y = h @ W2  (bf16 WMMA, f32 out) ----------------
__global__ __launch_bounds__(256)
void gemm2_kernel(const __bf16* __restrict__ h,     // [E,CAP,DF] bf16
                  const float* __restrict__ w2,     // [E,DF,DM] f32
                  float* __restrict__ y) {          // [E,CAP,DM] f32
    __shared__ __bf16 sA[64][32];
    __shared__ __bf16 sB[128][32];

    const int e     = blockIdx.z;
    const int m_blk = blockIdx.y * 64;              // over CAP
    const int n_blk = blockIdx.x * 128;             // over DM

    const __bf16* A = h  + (size_t)e * CAP * DF;
    const float*  B = w2 + (size_t)e * DF * DM;

    const int tid  = threadIdx.x;
    const int lane = tid & 31;
    const int half = lane >> 4;
    const int l16  = lane & 15;
    const int wm   = (tid >> 5) >> 2;
    const int wf   = (tid >> 5) & 3;

    const int a_row = tid >> 2;
    const int a_col = (tid & 3) * 8;
    const unsigned a_lds = (unsigned)(uintptr_t)&sA[a_row][a_col];

    v8f acc[2][2] = {};

    for (int k0 = 0; k0 < DF; k0 += 32) {
        __syncthreads();
        async_load_b128(a_lds, &A[(size_t)(m_blk + a_row) * DF + (k0 + a_col)]);
        #pragma unroll
        for (int it = 0; it < 4; ++it) {
            int i  = tid + it * 256;
            int kk = (i >> 6) * 2;
            int ff = (i & 63) * 2;
            const float2* pa = (const float2*)&B[(size_t)(k0 + kk)     * DM + (n_blk + ff)];
            const float2* pb = (const float2*)&B[(size_t)(k0 + kk + 1) * DM + (n_blk + ff)];
            float2 av = *pa, bv = *pb;
            *(unsigned*)&sB[ff][kk]     = pack_bf16(av.x, bv.x);
            *(unsigned*)&sB[ff + 1][kk] = pack_bf16(av.y, bv.y);
        }
        if ((k0 + 32 < DF) && (tid < 128)) {
            int kk = tid & 31, ff = (tid >> 5) * 32;
            __builtin_prefetch(&B[(size_t)(k0 + 32 + kk) * DM + (n_blk + ff)], 0, 1);
        }
        wait_async0();
        __syncthreads();

        #pragma unroll
        for (int i = 0; i < 2; ++i) {
            v16bf af;
            const int ar = wm * 32 + i * 16 + l16;
            #pragma unroll
            for (int j = 0; j < 16; ++j) af[j] = sA[ar][frag_k(j, half)];
            #pragma unroll
            for (int jj = 0; jj < 2; ++jj) {
                const int bc = wf * 32 + jj * 16 + l16;
                v16bf bfr;
                #pragma unroll
                for (int j = 0; j < 16; ++j) bfr[j] = sB[bc][frag_k(j, half)];
                acc[i][jj] = __builtin_amdgcn_wmma_f32_16x16x32_bf16(
                    false, af, false, bfr, (short)0, acc[i][jj], false, false);
            }
        }
    }

    float* Y = y + (size_t)e * CAP * DM;
    #pragma unroll
    for (int i = 0; i < 2; ++i)
        #pragma unroll
        for (int jj = 0; jj < 2; ++jj)
            #pragma unroll
            for (int r = 0; r < 8; ++r) {
                int row = m_blk + wm * 32 + i * 16 + r + (half << 3);
                int col = n_blk + wf * 32 + jj * 16 + l16;
                Y[(size_t)row * DM + col] = acc[i][jj][r];
            }
}

// ---------------- 6: combine (gather + gate-weight + sum over k) ----------------
__global__ __launch_bounds__(256)
void combine_kernel(const float* __restrict__ y, const int* __restrict__ slot,
                    const float* __restrict__ gate, float* __restrict__ out) {
    const int t = blockIdx.x;
    const int s0 = slot[2 * t], s1 = slot[2 * t + 1];
    const float g0 = gate[2 * t], g1 = gate[2 * t + 1];
    const bool k0 = s0 < ECAP, k1 = s1 < ECAP;
    const float* y0 = y + (size_t)(k0 ? s0 : 0) * DM;
    const float* y1 = y + (size_t)(k1 ? s1 : 0) * DM;
    for (int d = threadIdx.x; d < DM; d += 256) {
        float v = 0.f;
        if (k0) v += g0 * y0[d];
        if (k1) v += g1 * y1[d];
        out[(size_t)t * DM + d] = v;
    }
}

// ---------------- 7: aux loss (deterministic serial sum) ----------------
__global__ void aux_kernel(const float* __restrict__ blkImp,
                           const int* __restrict__ blkLoad,
                           float* __restrict__ out_aux) {
    if (blockIdx.x == 0 && threadIdx.x == 0) {
        float lb = 0.f;
        for (int e = 0; e < NE; ++e) {
            float imp = 0.f; int ld = 0;
            for (int b = 0; b < 32; ++b) {
                imp += blkImp[b * NE + e];
                ld  += blkLoad[b * NE + e];
            }
            lb += (imp / (float)T_TOK) * ((float)ld / (float)T_TOK);
        }
        out_aux[0] = 0.01f * (float)NE * lb;
    }
}

// ---------------- launch ----------------
extern "C" void kernel_launch(void* const* d_in, const int* in_sizes, int n_in,
                              void* d_out, int out_size, void* d_ws, size_t ws_size,
                              hipStream_t stream) {
    (void)in_sizes; (void)n_in; (void)out_size; (void)ws_size;
    const float* x  = (const float*)d_in[0];   // [4,2048,1024]
    const float* rw = (const float*)d_in[1];   // [1024,8]
    const float* rb = (const float*)d_in[2];   // [8]
    const float* w1 = (const float*)d_in[3];   // [8,1024,4096]
    const float* w3 = (const float*)d_in[4];   // [8,1024,4096]
    const float* w2 = (const float*)d_in[5];   // [8,4096,1024]
    float* out = (float*)d_out;                // [8192*1024] out + [1] aux

    // workspace layout (~225 MB)
    char* ws = (char*)d_ws;
    float*  gate    = (float*)(ws + 0);               //  64 KB
    int*    eidx    = (int*)  (ws + 65536);           //  64 KB
    int*    slot    = (int*)  (ws + 131072);          //  64 KB
    int*    invs    = (int*)  (ws + 196608);          //  64 KB
    float*  blkImp  = (float*)(ws + 262144);          //   1 KB
    int*    blkLoad = (int*)  (ws + 263168);          //   1 KB
    __bf16* buf     = (__bf16*)(ws + 264192);         //  32 MB  [E,CAP,DM]
    __bf16* hbuf    = (__bf16*)(ws + 264192 + (size_t)ECAP * DM * 2);   // 128 MB
    float*  ybuf    = (float*)(ws + 264192 + (size_t)ECAP * DM * 2
                                          + (size_t)ECAP * DF * 2);     //  64 MB

    init_kernel<<<ECAP / 256, 256, 0, stream>>>(invs);
    router_kernel<<<T_TOK / 256, 256, 0, stream>>>(x, rw, rb, gate, eidx, blkImp, blkLoad);
    scan_kernel<<<NE, 256, 0, stream>>>(eidx, slot, invs);
    gather_kernel<<<ECAP, 256, 0, stream>>>(x, invs, buf);
    gemm1_kernel<<<dim3(DF / 128, CAP / 64, NE), 256, 0, stream>>>(buf, w1, w3, hbuf);
    gemm2_kernel<<<dim3(DM / 128, CAP / 64, NE), 256, 0, stream>>>(hbuf, w2, ybuf);
    combine_kernel<<<T_TOK, 256, 0, stream>>>(ybuf, slot, gate, out);
    aux_kernel<<<1, 32, 0, stream>>>(blkImp, blkLoad, out + (size_t)T_TOK * DM);
}
